// RGCN_15006615732838
// MI455X (gfx1250) — matrix operand order repaired and checked by reading, not compiled, not run on hardware
//
#include <hip/hip_runtime.h>
#include <hip/hip_bf16.h>

typedef __attribute__((ext_vector_type(16))) _Float16 v16h;
typedef __attribute__((ext_vector_type(8)))  _Float16 v8h;
typedef __attribute__((ext_vector_type(8)))  float    v8f;

#define RG_N 50000
#define RG_E 800000
#define RG_R 8
#define RG_D 256

#if __has_builtin(__builtin_amdgcn_global_load_async_to_lds_b128) && \
    __has_builtin(__builtin_amdgcn_s_wait_asynccnt)
#define RG_ASYNC_LDS 1
typedef int rg_int4 __attribute__((ext_vector_type(4)));
typedef __attribute__((address_space(1))) rg_int4* rg_gptr;   // global src
typedef __attribute__((address_space(3))) rg_int4* rg_lptr;   // LDS dst
#endif

// ---------------------------------------------------------------------------
// f32 -> f16 convert
// ---------------------------------------------------------------------------
__global__ void rgcn_cvt_f16(const float* __restrict__ in,
                             _Float16* __restrict__ out, int n) {
    int i = blockIdx.x * 256 + threadIdx.x;
    if (i < n) out[i] = (_Float16)in[i];
}

// zero f32 buffer
__global__ void rgcn_zero_f32(float* __restrict__ p, int n) {
    int i = blockIdx.x * 256 + threadIdx.x;
    if (i < n) p[i] = 0.0f;
}

// ---------------------------------------------------------------------------
// Transpose + convert weights: W [R][K=256][N=256] f32 -> WT [R][N][K] f16.
// One-time 2MB job; lets the GEMM stage W slices into LDS with contiguous
// 16B chunks (async global->LDS capable).
// ---------------------------------------------------------------------------
__global__ void rgcn_transpose_w(const float* __restrict__ W,
                                 _Float16* __restrict__ WT, int total) {
    int idx = blockIdx.x * 256 + threadIdx.x;
    if (idx >= total) return;
    int r = idx >> 16;
    int k = (idx >> 8) & 255;
    int n = idx & 255;                 // coalesced read along n
    WT[((size_t)r * RG_D + n) * RG_D + k] = (_Float16)W[idx];
}

// ---------------------------------------------------------------------------
// WMMA GEMM:  C[M x 256] (f16) = A[M x 256] (f16) @ B[256 x 256]
// WT is B transposed: WT[n][k] f16 row-major.
// Block = 256 threads = 8 waves; wave w computes rows [blk*128 + 16w, +16) x
// all 256 cols (16 WMMA tiles). WT staged into LDS in 64-K slices via async
// global->LDS (ASYNCcnt) when available.
// ---------------------------------------------------------------------------
__global__ __launch_bounds__(256) void rgcn_gemm_wmma(
    const _Float16* __restrict__ A, const _Float16* __restrict__ WT,
    _Float16* __restrict__ C, int M) {
    // sB[n*72 + k_local], k_local in [0,64), row stride padded to 72 halves
    // (144B = 36 dwords -> conflict-free, 16B-aligned b128 accesses)
    __shared__ _Float16 sB[256 * 72];

    const int tid  = threadIdx.x;
    const int wave = tid >> 5;
    const int lane = tid & 31;
    const int c    = lane & 15;
    const int hi   = lane >> 4;

    const int  m16    = blockIdx.x * 128 + wave * 16;
    const bool active = (m16 < M);                 // wave-uniform
    const _Float16* aRow = A + (size_t)(m16 + c) * RG_D;

    v8f acc[16] = {};

    for (int s = 0; s < 4; ++s) {
        __syncthreads();
        // Stage WT[n][64s .. 64s+63] into LDS: 2048 contiguous 16B chunks.
#pragma unroll
        for (int i = 0; i < 8; ++i) {
            int idx = tid + 256 * i;       // 0..2047
            int n   = idx >> 3;
            int ch  = (idx & 7) * 8;       // half offset within 64-half row
            const _Float16* g = WT + (size_t)n * RG_D + s * 64 + ch;
            _Float16*       l = sB + n * 72 + ch;
#ifdef RG_ASYNC_LDS
            __builtin_amdgcn_global_load_async_to_lds_b128(
                (rg_gptr)(uintptr_t)g, (rg_lptr)(uintptr_t)l, 0, 0);
#else
            *(v8h*)l = *(const v8h*)g;
#endif
        }
#ifdef RG_ASYNC_LDS
        __builtin_amdgcn_s_wait_asynccnt(0);
#endif
        __syncthreads();

        if (active) {
#pragma unroll
            for (int kc = 0; kc < 2; ++kc) {
                const int kg = s * 64 + kc * 32;   // global K base of WMMA step
                // A fragment (16x32): lane half hi holds K = kg+8hi+0..7 and
                // kg+16+8hi+0..7 (CDNA5 A-matrix VGPR layout).
                v8h alo = *(const v8h*)(aRow + kg + hi * 8);
                v8h ahi = *(const v8h*)(aRow + kg + 16 + hi * 8);
                v16h a = __builtin_shufflevector(alo, ahi,
                    0,1,2,3,4,5,6,7,8,9,10,11,12,13,14,15);

                // B fragment: lane half hi holds K = 8hi+0..7 and 16+8hi+0..7
                const int kl = kc * 32 + hi * 8;
                const _Float16* bbase = sB + c * 72 + kl;
                v8h blo = *(const v8h*)(bbase);
                v8h bhi = *(const v8h*)(bbase + 16);
#pragma unroll
                for (int nt = 0; nt < 16; ++nt) {
                    v16h b = __builtin_shufflevector(blo, bhi,
                        0,1,2,3,4,5,6,7,8,9,10,11,12,13,14,15);
                    if (nt < 15) {   // software pipeline: prefetch next tile
                        const _Float16* bn = bbase + (nt + 1) * (16 * 72);
                        blo = *(const v8h*)(bn);
                        bhi = *(const v8h*)(bn + 16);
                    }
                    acc[nt] = __builtin_amdgcn_wmma_f32_16x16x32_f16(
                        false, a, false, b, (short)0, acc[nt], false, false);
                }
            }
        }
    }

    if (active) {
        // C/D layout: VGPR v, lane -> row m16 + v + 8*hi, col nt*16 + c
#pragma unroll
        for (int nt = 0; nt < 16; ++nt) {
#pragma unroll
            for (int v = 0; v < 8; ++v) {
                C[(size_t)(m16 + v + 8 * hi) * RG_D + nt * 16 + c] =
                    (_Float16)acc[nt][v];
            }
        }
    }
}

// ---------------------------------------------------------------------------
// Edge scatter for one relation: one wave per edge, lane handles 8 columns.
// msg = relu(xt[src] + bias); atomic-add into out[dst] (L2-resident).
// ---------------------------------------------------------------------------
__global__ __launch_bounds__(256) void rgcn_scatter(
    const _Float16* __restrict__ xt, const float* __restrict__ bias,
    const int* __restrict__ srcArr, const int* __restrict__ dstArr,
    const int* __restrict__ etArr, float* __restrict__ out, int rel, int E) {
    int e = blockIdx.x * 8 + (threadIdx.x >> 5);   // wave-uniform
    if (e >= E) return;
    if (etArr[e] != rel) return;
    int lane = threadIdx.x & 31;
    int src = srcArr[e];
    int dst = dstArr[e];

    v8h m = *(const v8h*)(xt + (size_t)src * RG_D + lane * 8);
    float*       o  = out  + (size_t)dst * RG_D + lane * 8;
    const float* bp = bias + lane * 8;
#pragma unroll
    for (int i = 0; i < 8; ++i) {
        float v = (float)m[i] + bp[i];
        v = v > 0.0f ? v : 0.0f;
        unsafeAtomicAdd(o + i, v);                 // global_atomic_add_f32
    }
}

// ---------------------------------------------------------------------------
// Launch
// ---------------------------------------------------------------------------
extern "C" void kernel_launch(void* const* d_in, const int* in_sizes, int n_in,
                              void* d_out, int out_size, void* d_ws, size_t ws_size,
                              hipStream_t stream) {
    const float* x  = (const float*)d_in[0];
    const float* W0 = (const float*)d_in[1];
    const float* b0 = (const float*)d_in[2];
    const float* W1 = (const float*)d_in[3];
    const float* b1 = (const float*)d_in[4];
    const int*   ei = (const int*)d_in[5];   // [2, E]
    const int*   et = (const int*)d_in[6];   // [E]
    const int* src = ei;
    const int* dst = ei + RG_E;

    const int ND = RG_N * RG_D;        // 12.8M
    const int WE = RG_R * RG_D * RG_D; // 524288 per layer

    // workspace layout (bytes):
    //   x16  : ND f16            = 25,600,000
    //   wT16 : 2*WE f16          =  2,097,152  (transposed weights, both layers)
    //   xt16 : ND f16            = 25,600,000  (one relation at a time, L2-resident)
    //   h    : ND f32            = 51,200,000
    char* ws = (char*)d_ws;
    _Float16* x16  = (_Float16*)ws;
    _Float16* wT16 = (_Float16*)(ws + 25600000);
    _Float16* xt16 = (_Float16*)(ws + 27697152);
    float*    h    = (float*)   (ws + 53297152);
    float*    out  = (float*)d_out;

    rgcn_cvt_f16<<<(ND + 255) / 256, 256, 0, stream>>>(x, x16, ND);
    rgcn_transpose_w<<<(WE + 255) / 256, 256, 0, stream>>>(W0, wT16, WE);
    rgcn_transpose_w<<<(WE + 255) / 256, 256, 0, stream>>>(W1, wT16 + WE, WE);
    rgcn_zero_f32<<<(ND + 255) / 256, 256, 0, stream>>>(h, ND);
    rgcn_zero_f32<<<(ND + 255) / 256, 256, 0, stream>>>(out, ND);

    const int gemmGrid    = (RG_N + 127) / 128;    // 391
    const int scatterGrid = (RG_E + 7) / 8;        // 100000

    for (int l = 0; l < 2; ++l) {
        const _Float16* Wl   = wT16 + l * WE;
        const float*    bias = l ? b1 : b0;
        float*          o    = l ? out : h;
        for (int r = 0; r < RG_R; ++r) {
            rgcn_gemm_wmma<<<gemmGrid, 256, 0, stream>>>(
                x16, Wl + r * RG_D * RG_D, xt16, RG_N);
            rgcn_scatter<<<scatterGrid, 256, 0, stream>>>(
                xt16, bias, src, dst, et, o, r, RG_E);
        }
        if (l == 0) {
            rgcn_cvt_f16<<<(ND + 255) / 256, 256, 0, stream>>>(h, x16, ND);
        }
    }
}
